// EncoderFLAX_89455578841392
// MI455X (gfx1250) — matrix operand order repaired and checked by reading, not compiled, not run on hardware
//
#include <hip/hip_runtime.h>
#include <hip/hip_bf16.h>
#include <math.h>

// Problem sizes (fixed by the reference).
#define B_ 256
#define S_ 256
#define E_ 1024
#define H_ 1024

// Tiling: each block computes a 64-row x (4 gates x 64 col) tile of z.
#define TM 64   // batch rows per block
#define TN 64   // columns per gate per block
#define KC 32   // K-chunk (bf16 WMMA depth)
#define NG 4    // gates (i,f,g,o)

typedef __attribute__((ext_vector_type(16))) __bf16 v16bf;
typedef __attribute__((ext_vector_type(8)))  float  v8f;

__device__ __forceinline__ float sigmoid_(float x) { return 1.0f / (1.0f + __expf(-x)); }

// Generic pointer -> LDS byte offset. Per CDNA5 ISA, an LDS-aperture generic
// address carries the wave-relative LDS offset in its low 32 bits.
__device__ __forceinline__ unsigned lds_off_(const void* p) {
    return (unsigned)(uintptr_t)p;
}

// CDNA5 async global->LDS copy of 64 contiguous bytes per lane (ASYNCcnt).
// INST_OFFSET is applied to BOTH the LDS and the global address, so all four
// b128 transfers share one LDS base VGPR and one global base VGPR pair.
__device__ __forceinline__ void async_copy_64B(unsigned ldsAddr, const void* gAddr) {
    asm volatile(
        "global_load_async_to_lds_b128 %0, %1, off\n\t"
        "global_load_async_to_lds_b128 %0, %1, off offset:16\n\t"
        "global_load_async_to_lds_b128 %0, %1, off offset:32\n\t"
        "global_load_async_to_lds_b128 %0, %1, off offset:48"
        :: "v"(ldsAddr), "v"(gAddr) : "memory");
}
__device__ __forceinline__ void wait_asynccnt0() {
    asm volatile("s_wait_asynccnt 0x0" ::: "memory");
}

// ---------------------------------------------------------------------------
// One-shot: transpose + convert W [K=1024][N=4096] fp32 -> Wt [N=4096][K=1024]
// bf16 so the step kernel's LDS weight tiles are contiguous 64B column chunks.
// grid = (K/64, N/64, 2); z=0 -> Wi, z=1 -> Wh.
// ---------------------------------------------------------------------------
__global__ __launch_bounds__(256) void transpose_cvt_kernel(
    const float* __restrict__ Wi, const float* __restrict__ Wh,
    __bf16* __restrict__ WtI, __bf16* __restrict__ WtH)
{
    const float* W  = (blockIdx.z == 0) ? Wi : Wh;
    __bf16*      Wt = (blockIdx.z == 0) ? WtI : WtH;

    __shared__ float tile[64][65];
    const int k0  = blockIdx.x * 64;
    const int n0  = blockIdx.y * 64;
    const int tid = threadIdx.x;

    const int tn = tid & 63;       // coalesced read along n
    const int tk = tid >> 6;       // 0..3
    #pragma unroll
    for (int i = 0; i < 16; ++i)
        tile[tk + i * 4][tn] = W[(size_t)(k0 + tk + i * 4) * (4 * H_) + n0 + tn];
    __syncthreads();

    const int wn = tid >> 2;            // 0..63 column within tile
    const int wk = (tid & 3) * 16;      // 0,16,32,48
    __bf16* dst = Wt + (size_t)(n0 + wn) * E_ + k0 + wk;
    #pragma unroll
    for (int i = 0; i < 16; ++i)
        dst[i] = (__bf16)tile[wk + i][wn];
}

// ---------------------------------------------------------------------------
// One LSTM timestep: z = x_t@Wi + h_{t-1}@Wh + b, then gates -> h_t, c_t.
// x_t gathered from emb; h_{t-1} read from out[:, t-1, :]; weights read from
// the pre-transposed bf16 copies via async global->LDS with double buffering.
// ---------------------------------------------------------------------------
__global__ __launch_bounds__(256) void lstm_step_kernel(
    const int*    __restrict__ inputs,  // [B,S]
    const float*  __restrict__ emb,     // [V,E]
    const __bf16* __restrict__ WtI,     // [4H][E]  bf16, k-contiguous
    const __bf16* __restrict__ WtH,     // [4H][H]  bf16, k-contiguous
    const float*  __restrict__ bias,    // [4H]
    float*        __restrict__ out,     // [B,S,H]
    float*        __restrict__ cbuf,    // [B,H] cell state (in-place)
    int t)
{
    const int m0   = blockIdx.x * TM;
    const int n0   = blockIdx.y * TN;
    const int tid  = threadIdx.x;
    const int lane = tid & 31;
    const int wave = tid >> 5;          // 0..7
    const int r    = wave & 1;          // row half: rows [r*32, r*32+32)
    const int nt   = wave >> 1;         // 16-wide n sub-tile 0..3
    const bool first = (t == 0);

    __shared__ __align__(16) __bf16 Alds[2][TM * KC];        // 2 x 4 KB
    __shared__ __align__(16) __bf16 Blds[2][NG * TN * KC];   // 2 x 16 KB

    // Accumulators acc[mt][gate], initialized with bias (depends on n only).
    v8f acc[2][4];
    {
        const int ncol = n0 + nt * 16 + (lane & 15);
        #pragma unroll
        for (int g = 0; g < 4; ++g) {
            const float bv = bias[g * H_ + ncol];
            v8f a;
            #pragma unroll
            for (int v = 0; v < 8; ++v) a[v] = bv;
            acc[0][g] = a;
            acc[1][g] = a;
        }
    }

    // Staging work distribution (256 threads).
    const int aRow = tid >> 2;          // 0..63 A row
    const int aK   = (tid & 3) * 8;     // 0,8,16,24
    const int embIdx = inputs[(m0 + aRow) * S_ + t];
    const int bg = tid >> 6;            // gate
    const int bn = tid & 63;            // column within gate tile
    const size_t bColBase = (size_t)(bg * H_ + n0 + bn) * E_;  // row of Wt (len 1024)

    const int kTotal  = first ? E_ : (E_ + H_);  // h_{-1}==0: skip Wh phase at t=0
    const int nChunks = kTotal / KC;

    auto stage = [&](int k0, int buf) {
        const bool xph = (k0 < E_);
        // ---- B: async bf16 copy, 64B per thread (one weight column chunk) ----
        const __bf16* bsrc = (xph ? WtI : WtH) + bColBase + (xph ? k0 : k0 - E_);
        async_copy_64B(lds_off_(&Blds[buf][tid * KC]), bsrc);
        // ---- A: fp32 load + cvt + ds_store (8 elements per thread) ----
        const float* asrc = xph
            ? (emb + (size_t)embIdx * E_ + (k0 + aK))
            : (out + ((size_t)(m0 + aRow) * S_ + (t - 1)) * H_ + (k0 - E_ + aK));
        const float4 f0 = *(const float4*)asrc;
        const float4 f1 = *(const float4*)(asrc + 4);
        __bf16* al = &Alds[buf][aRow * KC + aK];
        al[0] = (__bf16)f0.x; al[1] = (__bf16)f0.y; al[2] = (__bf16)f0.z; al[3] = (__bf16)f0.w;
        al[4] = (__bf16)f1.x; al[5] = (__bf16)f1.y; al[6] = (__bf16)f1.z; al[7] = (__bf16)f1.w;
    };

    // Prologue: fill buffer 0.
    stage(0, 0);
    wait_asynccnt0();
    __syncthreads();

    for (int c = 0; c < nChunks; ++c) {
        const int cur = c & 1, nxt = cur ^ 1;
        // Next chunk's loads in flight while we compute the current chunk.
        // Safe: the barrier ending iteration c-1 guarantees all waves finished
        // reading buf[nxt] (it was 'cur' then).
        if (c + 1 < nChunks) stage((c + 1) * KC, nxt);

        // ---- A fragments: lane holds K contiguous {kb..kb+15}, kb=(lane/16)*16 ----
        v16bf afr[2];
        {
            const int ar = lane & 15;
            const int kb = (lane >> 4) * 16;
            #pragma unroll
            for (int mt = 0; mt < 2; ++mt) {
                const __bf16* p = &Alds[cur][(r * 32 + mt * 16 + ar) * KC];
                uint4* d = reinterpret_cast<uint4*>(&afr[mt]);
                d[0] = *reinterpret_cast<const uint4*>(p + kb);
                d[1] = *reinterpret_cast<const uint4*>(p + kb + 8);
            }
        }
        // ---- B fragments + WMMA: lane holds col=lane%16, K contiguous (lane/16)*16 ----
        #pragma unroll
        for (int g = 0; g < 4; ++g) {
            v16bf bfr;
            const __bf16* p =
                &Blds[cur][((g * TN) + nt * 16 + (lane & 15)) * KC + (lane >> 4) * 16];
            uint4* d = reinterpret_cast<uint4*>(&bfr);
            d[0] = reinterpret_cast<const uint4*>(p)[0];
            d[1] = reinterpret_cast<const uint4*>(p)[1];
            #pragma unroll
            for (int mt = 0; mt < 2; ++mt)
                acc[mt][g] = __builtin_amdgcn_wmma_f32_16x16x32_bf16(
                    false, afr[mt], false, bfr, (short)0, acc[mt][g], false, false);
        }
        wait_asynccnt0();   // my async writes into buf[nxt] must land before barrier
        __syncthreads();
    }

    // ---- gate math: this wave owns all 4 gates for its (m,n) footprint ----
    const int colH = n0 + nt * 16 + (lane & 15);
    #pragma unroll
    for (int mt = 0; mt < 2; ++mt) {
        const int mBase = m0 + r * 32 + mt * 16 + (lane >> 4) * 8;
        #pragma unroll
        for (int v = 0; v < 8; ++v) {
            const int row = mBase + v;
            const size_t cIx = (size_t)row * H_ + colH;
            const float zi = acc[mt][0][v];
            const float zf = acc[mt][1][v];
            const float zg = acc[mt][2][v];
            const float zo = acc[mt][3][v];
            const float cOld = first ? 0.0f : cbuf[cIx];
            const float cNew = sigmoid_(zf) * cOld + sigmoid_(zi) * tanhf(zg);
            const float h    = sigmoid_(zo) * tanhf(cNew);
            cbuf[cIx] = cNew;
            out[((size_t)row * S_ + t) * H_ + colH] = h;
        }
    }
}

// Copy final h (= outputs[:, S-1, :], strided) and c into their d_out slots.
__global__ __launch_bounds__(256) void lstm_finalize_kernel(
    const float* __restrict__ out,
    const float* __restrict__ cbuf,
    float*       __restrict__ hOut,
    float*       __restrict__ cOut)
{
    const int i = blockIdx.x * blockDim.x + threadIdx.x;  // 0 .. B*H-1
    const int b = i / H_;
    const int h = i % H_;
    hOut[i] = out[((size_t)b * S_ + (S_ - 1)) * H_ + h];
    cOut[i] = cbuf[i];
}

extern "C" void kernel_launch(void* const* d_in, const int* in_sizes, int n_in,
                              void* d_out, int out_size, void* d_ws, size_t ws_size,
                              hipStream_t stream)
{
    const int*   inputs = (const int*)  d_in[0];  // [B,S] int32
    const float* emb    = (const float*)d_in[1];  // [V,E]
    const float* Wi     = (const float*)d_in[2];  // [E,4H]
    const float* Wh     = (const float*)d_in[3];  // [H,4H]
    const float* bias   = (const float*)d_in[4];  // [4H]

    float* out  = (float*)d_out;                  // outputs [B,S,H]
    float* hOut = out + (size_t)B_ * S_ * H_;     // final h [B,H]
    float* cOut = hOut + (size_t)B_ * H_;         // final c [B,H]

    // Workspace layout: c state (1 MB) | WtI bf16 (8 MB) | WtH bf16 (8 MB).
    char*   ws   = (char*)d_ws;
    float*  cbuf = (float*)ws;
    __bf16* WtI  = (__bf16*)(ws + (size_t)B_ * H_ * sizeof(float));
    __bf16* WtH  = WtI + (size_t)(4 * H_) * E_;

    // One-shot weight transpose+convert (re-done every call: deterministic).
    dim3 tgrid(E_ / 64, (4 * H_) / 64, 2);        // (16, 64, 2)
    transpose_cvt_kernel<<<tgrid, dim3(256), 0, stream>>>(Wi, Wh, WtI, WtH);

    // Sequential scan: 256 dependent step kernels on the stream.
    dim3 grid(B_ / TM, H_ / TN);                  // (4, 16) = 64 blocks
    for (int t = 0; t < S_; ++t)
        lstm_step_kernel<<<grid, dim3(256), 0, stream>>>(
            inputs, emb, WtI, WtH, bias, out, cbuf, t);

    lstm_finalize_kernel<<<(B_ * H_) / 256, 256, 0, stream>>>(out, cbuf, hOut, cOut);
}
